// Moondream3TextMoE_25975962206337
// MI455X (gfx1250) — matrix operand order repaired and checked by reading, not compiled, not run on hardware
//
#include <hip/hip_runtime.h>
#include <cstdint>
#include <cstddef>

// Problem constants (from reference setup_inputs)
#define T_TOK 4096
#define HDIM  2048
#define IDIM  1024
#define NEXP  64
#define TOPK  8

// Tiling
#define TM   256      // gathered rows per block
#define KC   32       // K chunk (one bf16 WMMA K-step)
#define LDK  40       // padded LDS row stride (elements) to dodge bank conflicts
#define TNA  32       // fc1: I-columns per block
#define TNB  64       // fc2: H-columns per block

typedef __attribute__((ext_vector_type(16))) __bf16          v16bf;
typedef __attribute__((ext_vector_type(8)))  float           v8f;
typedef __attribute__((ext_vector_type(8)))  unsigned short  u16x8;
typedef __attribute__((ext_vector_type(16))) unsigned short  u16x16;

__device__ __forceinline__ unsigned short f2bf(float f) {
  // round-to-nearest-even fp32 -> bf16
  unsigned int u = __builtin_bit_cast(unsigned int, f);
  u += 0x7FFFu + ((u >> 16) & 1u);
  return (unsigned short)(u >> 16);
}

__device__ __forceinline__ v8f zero8() {
  v8f z = {0.f, 0.f, 0.f, 0.f, 0.f, 0.f, 0.f, 0.f};
  return z;
}

// Low 32 bits of a generic pointer to __shared__ = wave-relative LDS offset
__device__ __forceinline__ unsigned lds_off(const void* p) {
  return (unsigned)(uintptr_t)p;
}

// Async global->LDS copy, 16B per lane (VGLOBAL op 98, tracked by ASYNCcnt)
__device__ __forceinline__ void async_ld_b128(unsigned lds, const void* g) {
  asm volatile("global_load_async_to_lds_b128 %0, %1, off"
               :: "v"(lds), "v"(g) : "memory");
}

__device__ __forceinline__ void wait_async0() {
  asm volatile("s_wait_asynccnt 0" ::: "memory");
}

// A fragment (16x32 bf16, MxK): lane = M (lane&15); half = lane>>4 selects
// K chunks {8h..8h+7} and {16+8h..16+8h+7} (ISA 7.12.2, 16-bit A 16x32).
__device__ __forceinline__ v16bf frag_a(const unsigned short* rowp, int half) {
  u16x8 lo = *(const u16x8*)(rowp + half * 8);
  u16x8 hi = *(const u16x8*)(rowp + 16 + half * 8);
  u16x16 c;
#pragma unroll
  for (int i = 0; i < 8; ++i) { c[i] = lo[i]; c[8 + i] = hi[i]; }
  return __builtin_bit_cast(v16bf, c);
}

// B fragment (32x16 bf16, KxN), supplied as W rows (N-major): lane = N;
// half = lane>>4 selects K chunks {8h..8h+7} and {16+8h..16+8h+7}.
__device__ __forceinline__ v16bf frag_b(const unsigned short* rowp, int half) {
  u16x8 lo = *(const u16x8*)(rowp + half * 8);
  u16x8 hi = *(const u16x8*)(rowp + 16 + half * 8);
  u16x16 c;
#pragma unroll
  for (int i = 0; i < 8; ++i) { c[i] = lo[i]; c[8 + i] = hi[i]; }
  return __builtin_bit_cast(v16bf, c);
}

__device__ __forceinline__ v8f wmma_bf16(v16bf a, v16bf b, v8f c) {
  return __builtin_amdgcn_wmma_f32_16x16x32_bf16(false, a, false, b,
                                                 (short)0, c, false, false);
}

// ---------------------------------------------------------------------------
// Kernel 0: zero per-expert counters + zero pad page (ws not re-poisoned)
// ---------------------------------------------------------------------------
__global__ void moe_init_kernel(int* __restrict__ counts,
                                unsigned int* __restrict__ zpad) {
  counts[threadIdx.x] = 0;
  zpad[threadIdx.x] = 0u;  // 64 threads -> 256B of zeros
}

// ---------------------------------------------------------------------------
// Kernel 1: router — logits, top-8, softmax, expert lists, x->bf16, out=0
// ---------------------------------------------------------------------------
__global__ __launch_bounds__(256) void moe_router_kernel(
    const float* __restrict__ x, const float* __restrict__ gate_w,
    const float* __restrict__ gate_b, float* __restrict__ comb,
    int* __restrict__ counts, int* __restrict__ lists,
    unsigned short* __restrict__ xb, float* __restrict__ out) {
  const int t = blockIdx.x;
  const int tid = threadIdx.x;

  __shared__ float xrow[HDIM];
  __shared__ float partial[256];
  __shared__ float logit[NEXP];

  for (int i = tid; i < HDIM; i += 256) {
    float v = x[(size_t)t * HDIM + i];
    xrow[i] = v;
    xb[(size_t)t * HDIM + i] = f2bf(v);
    out[(size_t)t * HDIM + i] = 0.0f;   // zero output (stage-B atomics add)
  }
  __syncthreads();

  {  // 4 threads per expert
    const int e = tid >> 2, s = tid & 3;
    const float* w = gate_w + (size_t)e * HDIM;
    float sum = 0.f;
    for (int i = s; i < HDIM; i += 4) sum += xrow[i] * w[i];
    partial[tid] = sum;
  }
  __syncthreads();
  if (tid < NEXP) {
    logit[tid] = partial[tid * 4] + partial[tid * 4 + 1] +
                 partial[tid * 4 + 2] + partial[tid * 4 + 3] + gate_b[tid];
  }
  __syncthreads();

  if (tid == 0) {
    int   sidx[TOPK];
    float sval[TOPK];
#pragma unroll
    for (int k = 0; k < TOPK; ++k) {
      int bi = 0; float bv = -1e30f;
      for (int e = 0; e < NEXP; ++e)
        if (logit[e] > bv) { bv = logit[e]; bi = e; }
      sidx[k] = bi; sval[k] = bv; logit[bi] = -1e30f;
    }
    const float mx = sval[0];
    float ex[TOPK], den = 0.f;
#pragma unroll
    for (int k = 0; k < TOPK; ++k) { ex[k] = __expf(sval[k] - mx); den += ex[k]; }
    const float inv = 1.0f / den;
#pragma unroll
    for (int k = 0; k < TOPK; ++k) {
      const int e = sidx[k];
      comb[(size_t)t * NEXP + e] = ex[k] * inv;
      const int pos = atomicAdd(&counts[e], 1);
      lists[(size_t)e * T_TOK + pos] = t;
    }
  }
}

// ---------------------------------------------------------------------------
// Kernel 2: exclusive prefix sum of counts -> activation base per expert
// ---------------------------------------------------------------------------
__global__ void moe_prefix_kernel(const int* __restrict__ counts,
                                  int* __restrict__ pbase) {
  int acc = 0;
  for (int e = 0; e < NEXP; ++e) { pbase[e] = acc; acc += counts[e]; }
  pbase[NEXP] = acc;
}

// ---------------------------------------------------------------------------
// Kernel 3: fc1 + GeGLU (gathered rows). Double-buffered LDS; X tiles via
// GLOBAL_LOAD_ASYNC_TO_LDS, W1 tiles via register prefetch + fp32->bf16.
// grid: (E, T/TM, I/TNA); block: 256 (8 waves). Each wave: 32 rows x 32 cols.
// ---------------------------------------------------------------------------
__global__ __launch_bounds__(256) void moe_fc1_kernel(
    const float* __restrict__ fc1_w, const unsigned short* __restrict__ xb,
    const int* __restrict__ lists, const int* __restrict__ counts,
    const int* __restrict__ pbase, const unsigned short* __restrict__ zpad,
    unsigned short* __restrict__ act) {
  const int e = blockIdx.x;
  const int tokBase = blockIdx.y * TM;
  const int j0 = blockIdx.z * TNA;
  const int cnt = counts[e];
  if (tokBase >= cnt) return;  // block-uniform: EXEC all-ones for WMMA

  __shared__ unsigned short Xs[2][TM][LDK];
  __shared__ unsigned short Ws[2][2 * TNA][LDK];

  const int tid = threadIdx.x;
  const int lane = tid & 31;
  const int wave = tid >> 5;
  const int half = lane >> 4;
  const int lrow = lane & 15;

  // My gathered source row (rows past cnt async-load from the zero pad).
  const int i_row = tokBase + tid;
  const unsigned short* xrow =
      (i_row < cnt) ? (xb + (size_t)lists[(size_t)e * T_TOK + i_row] * HDIM)
                    : (const unsigned short*)nullptr;

  const float* W1 = fc1_w + (size_t)e * (2 * IDIM) * HDIM;
  const int wr = tid >> 2;
  const int wc = (tid & 3) * 8;
  const int wrow = (wr < TNA) ? (j0 + wr) : (IDIM + j0 + (wr - TNA));
  const float* wsrc = W1 + (size_t)wrow * HDIM + wc;

  float4 wf0, wf1;

#define ISSUE_X(K0, B) do {                                         \
    const unsigned short* g_ = xrow ? (xrow + (K0)) : zpad;         \
    unsigned l_ = lds_off(&Xs[(B)][tid][0]);                        \
    async_ld_b128(l_, g_);                                          \
    async_ld_b128(l_ + 16, g_ + 8);                                 \
    async_ld_b128(l_ + 32, g_ + 16);                                \
    async_ld_b128(l_ + 48, g_ + 24);                                \
  } while (0)
#define LOAD_W(K0) do {                                             \
    const float* s_ = wsrc + (K0);                                  \
    wf0 = *(const float4*)(s_);                                     \
    wf1 = *(const float4*)(s_ + 4);                                 \
    if ((K0) + KC < HDIM)                                           \
      __builtin_prefetch((const void*)(s_ + KC), 0, 1);             \
  } while (0)
#define STORE_W(B) do {                                             \
    unsigned short* d_ = &Ws[(B)][wr][wc];                          \
    d_[0] = f2bf(wf0.x); d_[1] = f2bf(wf0.y);                       \
    d_[2] = f2bf(wf0.z); d_[3] = f2bf(wf0.w);                       \
    d_[4] = f2bf(wf1.x); d_[5] = f2bf(wf1.y);                       \
    d_[6] = f2bf(wf1.z); d_[7] = f2bf(wf1.w);                       \
  } while (0)

  v8f acc_h[2][2], acc_g[2][2];
#pragma unroll
  for (int a = 0; a < 2; ++a)
#pragma unroll
    for (int b = 0; b < 2; ++b) { acc_h[a][b] = zero8(); acc_g[a][b] = zero8(); }

  // Pipeline prologue: tile 0 into buffer 0.
  ISSUE_X(0, 0);
  LOAD_W(0);
  STORE_W(0);
  wait_async0();
  __syncthreads();

  for (int k0 = 0; k0 < HDIM; k0 += KC) {
    const int cur = (k0 / KC) & 1;
    const int nxt = cur ^ 1;
    const bool more = (k0 + KC) < HDIM;
    if (more) { ISSUE_X(k0 + KC, nxt); LOAD_W(k0 + KC); }  // overlap w/ WMMA

#pragma unroll
    for (int rt = 0; rt < 2; ++rt) {
      const int row = wave * 32 + rt * 16 + lrow;
      const v16bf a = frag_a(&Xs[cur][row][0], half);
#pragma unroll
      for (int ct = 0; ct < 2; ++ct) {
        const v16bf bh = frag_b(&Ws[cur][ct * 16 + lrow][0], half);
        acc_h[rt][ct] = wmma_bf16(a, bh, acc_h[rt][ct]);
        const v16bf bg = frag_b(&Ws[cur][TNA + ct * 16 + lrow][0], half);
        acc_g[rt][ct] = wmma_bf16(a, bg, acc_g[rt][ct]);
      }
    }

    if (more) STORE_W(nxt);
    wait_async0();     // my async writes to LDS[nxt] landed
    __syncthreads();   // (compiler adds dscnt wait for my ds_stores)
  }
#undef ISSUE_X
#undef LOAD_W
#undef STORE_W

  const int pb = pbase[e];
#pragma unroll
  for (int rt = 0; rt < 2; ++rt) {
#pragma unroll
    for (int ct = 0; ct < 2; ++ct) {
      const v8f hv = acc_h[rt][ct];
      const v8f gv = acc_g[rt][ct];
#pragma unroll
      for (int r = 0; r < 8; ++r) {
        const int m = wave * 32 + rt * 16 + r + half * 8;
        const int i = tokBase + m;
        if (i < cnt) {
          const float h = hv[r], g = gv[r];
          const float a = 0.5f * h * (1.0f + erff(h * 0.70710678118f)) * (g + 1.0f);
          act[(size_t)(pb + i) * IDIM + (j0 + ct * 16 + lrow)] = f2bf(a);
        }
      }
    }
  }
}

// ---------------------------------------------------------------------------
// Kernel 4: fc2 + scaled scatter-add into out. Same double-buffered pipeline.
// grid: (E, T/TM, H/TNB); block: 256. Each wave: 32 rows x 64 cols.
// ---------------------------------------------------------------------------
__global__ __launch_bounds__(256) void moe_fc2_kernel(
    const float* __restrict__ fc2_w, const unsigned short* __restrict__ act,
    const int* __restrict__ lists, const int* __restrict__ counts,
    const int* __restrict__ pbase, const float* __restrict__ comb,
    const unsigned short* __restrict__ zpad, float* __restrict__ out) {
  const int e = blockIdx.x;
  const int tokBase = blockIdx.y * TM;
  const int n0 = blockIdx.z * TNB;
  const int cnt = counts[e];
  if (tokBase >= cnt) return;

  __shared__ unsigned short As[2][TM][LDK];
  __shared__ unsigned short Ws[2][TNB][LDK];
  __shared__ int   tokIds[TM];
  __shared__ float cws[TM];

  const int tid = threadIdx.x;
  const int lane = tid & 31;
  const int wave = tid >> 5;
  const int half = lane >> 4;
  const int lrow = lane & 15;
  const int pb = pbase[e];

  const int i_row = tokBase + tid;
  {
    const int tok = (i_row < cnt) ? lists[(size_t)e * T_TOK + i_row] : 0;
    tokIds[tid] = tok;
    cws[tid] = (i_row < cnt) ? comb[(size_t)tok * NEXP + e] : 0.f;
  }
  const unsigned short* arow =
      (i_row < cnt) ? (act + (size_t)(pb + i_row) * IDIM)
                    : (const unsigned short*)nullptr;

  const float* W2 = fc2_w + (size_t)e * HDIM * IDIM;
  const int wr = tid >> 2;
  const int wc = (tid & 3) * 8;
  const float* wsrc = W2 + (size_t)(n0 + wr) * IDIM + wc;

  float4 wf0, wf1;

#define ISSUE_A(K0, B) do {                                         \
    const unsigned short* g_ = arow ? (arow + (K0)) : zpad;         \
    unsigned l_ = lds_off(&As[(B)][tid][0]);                        \
    async_ld_b128(l_, g_);                                          \
    async_ld_b128(l_ + 16, g_ + 8);                                 \
    async_ld_b128(l_ + 32, g_ + 16);                                \
    async_ld_b128(l_ + 48, g_ + 24);                                \
  } while (0)
#define LOAD_W(K0) do {                                             \
    const float* s_ = wsrc + (K0);                                  \
    wf0 = *(const float4*)(s_);                                     \
    wf1 = *(const float4*)(s_ + 4);                                 \
    if ((K0) + KC < IDIM)                                           \
      __builtin_prefetch((const void*)(s_ + KC), 0, 1);             \
  } while (0)
#define STORE_W(B) do {                                             \
    unsigned short* d_ = &Ws[(B)][wr][wc];                          \
    d_[0] = f2bf(wf0.x); d_[1] = f2bf(wf0.y);                       \
    d_[2] = f2bf(wf0.z); d_[3] = f2bf(wf0.w);                       \
    d_[4] = f2bf(wf1.x); d_[5] = f2bf(wf1.y);                       \
    d_[6] = f2bf(wf1.z); d_[7] = f2bf(wf1.w);                       \
  } while (0)

  v8f acc[2][4];
#pragma unroll
  for (int a = 0; a < 2; ++a)
#pragma unroll
    for (int b = 0; b < 4; ++b) acc[a][b] = zero8();

  ISSUE_A(0, 0);
  LOAD_W(0);
  STORE_W(0);
  wait_async0();
  __syncthreads();  // also publishes tokIds/cws

  for (int k0 = 0; k0 < IDIM; k0 += KC) {
    const int cur = (k0 / KC) & 1;
    const int nxt = cur ^ 1;
    const bool more = (k0 + KC) < IDIM;
    if (more) { ISSUE_A(k0 + KC, nxt); LOAD_W(k0 + KC); }

#pragma unroll
    for (int rt = 0; rt < 2; ++rt) {
      const int row = wave * 32 + rt * 16 + lrow;
      const v16bf a = frag_a(&As[cur][row][0], half);
#pragma unroll
      for (int ct = 0; ct < 4; ++ct) {
        const v16bf b = frag_b(&Ws[cur][ct * 16 + lrow][0], half);
        acc[rt][ct] = wmma_bf16(a, b, acc[rt][ct]);
      }
    }

    if (more) STORE_W(nxt);
    wait_async0();
    __syncthreads();
  }
#undef ISSUE_A
#undef LOAD_W
#undef STORE_W

#pragma unroll
  for (int rt = 0; rt < 2; ++rt) {
#pragma unroll
    for (int ct = 0; ct < 4; ++ct) {
      const v8f yv = acc[rt][ct];
#pragma unroll
      for (int r = 0; r < 8; ++r) {
        const int m = wave * 32 + rt * 16 + r + half * 8;
        const int i = tokBase + m;
        if (i < cnt) {
          atomicAdd(out + (size_t)tokIds[m] * HDIM + (n0 + ct * 16 + lrow),
                    cws[m] * yv[r]);
        }
      }
    }
  }
}

// ---------------------------------------------------------------------------
extern "C" void kernel_launch(void* const* d_in, const int* in_sizes, int n_in,
                              void* d_out, int out_size, void* d_ws,
                              size_t ws_size, hipStream_t stream) {
  (void)in_sizes; (void)n_in; (void)out_size; (void)ws_size;
  const float* x      = (const float*)d_in[0];
  const float* gate_w = (const float*)d_in[1];
  const float* gate_b = (const float*)d_in[2];
  const float* fc1_w  = (const float*)d_in[3];
  const float* fc2_w  = (const float*)d_in[4];
  float* out = (float*)d_out;

  // Workspace layout (all 256B-aligned):
  char* w = (char*)d_ws;
  size_t off = 0;
  auto take = [&](size_t bytes) {
    size_t o = off;
    off = (off + bytes + 255) & ~(size_t)255;
    return o;
  };
  int*  counts = (int*)(w + take(NEXP * sizeof(int)));
  int*  pbase  = (int*)(w + take((NEXP + 1) * sizeof(int)));
  unsigned int* zpad = (unsigned int*)(w + take(256));
  float* comb  = (float*)(w + take((size_t)T_TOK * NEXP * sizeof(float)));
  int*  lists  = (int*)(w + take((size_t)NEXP * T_TOK * sizeof(int)));
  unsigned short* xb  = (unsigned short*)(w + take((size_t)T_TOK * HDIM * 2));
  unsigned short* act = (unsigned short*)(w + take((size_t)T_TOK * TOPK * IDIM * 2));

  moe_init_kernel<<<1, NEXP, 0, stream>>>(counts, zpad);
  moe_router_kernel<<<T_TOK, 256, 0, stream>>>(x, gate_w, gate_b, comb, counts,
                                               lists, xb, out);
  moe_prefix_kernel<<<1, 1, 0, stream>>>(counts, pbase);
  moe_fc1_kernel<<<dim3(NEXP, T_TOK / TM, IDIM / TNA), 256, 0, stream>>>(
      fc1_w, xb, lists, counts, pbase, (const unsigned short*)zpad, act);
  moe_fc2_kernel<<<dim3(NEXP, T_TOK / TM, HDIM / TNB), 256, 0, stream>>>(
      fc2_w, act, lists, counts, pbase, comb, (const unsigned short*)zpad, out);
}
